// ANIMATwo_86887188398378
// MI455X (gfx1250) — matrix operand order repaired and checked by reading, not compiled, not run on hardware
//
#include <hip/hip_runtime.h>
#include <hip/hip_bf16.h>
#include <math.h>

// ---------------------------------------------------------------------------
// Model dims (fixed by reference)
// ---------------------------------------------------------------------------
#define BB   64
#define TT   4096
#define DST  32
#define NC   1024          // TT / CHUNK
#define NBT  4             // batch tiles of 16

typedef __attribute__((ext_vector_type(16))) _Float16 v16h;
typedef __attribute__((ext_vector_type(8)))  float    v8f;

__device__ __forceinline__ v8f wmma_f16(v16h a, v16h b, v8f c) {
  // D = A(16x32 f16) * B(32x16 f16) + C(16x16 f32)
  return __builtin_amdgcn_wmma_f32_16x16x32_f16(false, a, false, b, (short)0, c, false, false);
}

// Fast reciprocal: single v_rcp_f32 instead of the precise-division macro
// sequence (v_div_scale/v_div_fmas/v_div_fixup) the compiler emits for '/'.
__device__ __forceinline__ float fast_rcp(float x) {
#if __has_builtin(__builtin_amdgcn_rcpf)
  return __builtin_amdgcn_rcpf(x);
#else
  return 1.f / x;
#endif
}
// sigmoid: v_exp + v_add + v_rcp  (short dependent chain for the serial scan)
__device__ __forceinline__ float sigm(float x) {
  return fast_rcp(1.f + __expf(-x));
}
// tanh(x) = 1 - 2/(e^{2x}+1): v_exp + v_add + v_rcp + fma
__device__ __forceinline__ float fast_tanh(float x) {
  return 1.f - 2.f * fast_rcp(__expf(2.f * x) + 1.f);
}

// ---------------------------------------------------------------------------
// Input order (setup_inputs dict insertion order, params flattened)
// ---------------------------------------------------------------------------
enum {
  IN_X = 0, IN_ENC_W, IN_ENC_B, IN_PROJ_W, IN_PROJ_B,
  IN_WZ_W, IN_WZ_B, IN_WR_W, IN_WR_B, IN_WH_W, IN_WH_B,
  IN_SFS, IN_SFM, IN_SFD,
  IN_MZ_W, IN_MZ_B, IN_MR_W, IN_MR_B, IN_MH_W, IN_MH_B,
  IN_DFS, IN_DFM, IN_DFD,
  IN_PHIS_W, IN_PHIS_B, IN_PHID_W, IN_PHID_B,
  IN_RES, IN_QKV_W, IN_QKV_B, IN_OUT_W, IN_OUT_B,
  IN_LN_G, IN_LN_B, IN_F1_W, IN_F1_B, IN_F2_W, IN_F2_B,
  IN_COMP_W, IN_COMP_B, IN_EXP_W, IN_EXP_B, IN_HEAD_W, IN_HEAD_B
};

// ---------------------------------------------------------------------------
// Workspace layout (bytes)
// ---------------------------------------------------------------------------
#define OFF_HPK     ((size_t)0)                       // f16, 4*4096*32*16 = 16 MB
#define OFF_STATES  ((size_t)16777216)                // f32 packed,       32 MB
#define OFF_FINALS  (OFF_STATES + (size_t)33554432)   // f32 [b][c][32]     8 MB
#define OFF_SUMM    (OFF_FINALS + (size_t)8388608)    // f32 [b][c][32]     8 MB
#define OFF_QKV     (OFF_SUMM   + (size_t)8388608)    // f32 [b][c][96]    24 MB
#define OFF_ATT     (OFF_QKV    + (size_t)25165824)   // f32 [b][c][32]     8 MB
#define OFF_XGF     (OFF_ATT    + (size_t)8388608)    // f32 [b][c][4]      1 MB
#define OFF_WFP     (OFF_XGF    + (size_t)1048576)    // f32 32x8
#define OFF_BFP     (OFF_WFP    + (size_t)1024)       // f32 32
#define OFF_GRUPK   (OFF_BFP    + (size_t)256)        // f16 3*2*2*512 = 12 KB
#define OFF_WFPK    (OFF_GRUPK  + (size_t)12288)      // f16 512
#define OFF_WF      (OFF_WFPK   + (size_t)1024)       // f32 4x32
#define OFF_BF      (OFF_WF     + (size_t)512)        // f32 4

// ===========================================================================
// Prep: fuse affine chains, pack weights into WMMA B-fragment f16 layouts.
// B-frag (16-bit, 32x16): lane l, elem i -> k = (i>=8?16:0)+(l>=16?8:0)+(i&7), n = l&15.
// ===========================================================================
__global__ __launch_bounds__(512) void prep_kernel(
    const float* __restrict__ enc_W, const float* __restrict__ enc_b,
    const float* __restrict__ proj_W, const float* __restrict__ proj_b,
    const float* __restrict__ Wz_W, const float* __restrict__ Wr_W,
    const float* __restrict__ Wh_W,
    const float* __restrict__ comp_W, const float* __restrict__ comp_b,
    const float* __restrict__ exp_W,  const float* __restrict__ exp_b,
    const float* __restrict__ head_W, const float* __restrict__ head_b,
    float* __restrict__ Wfp, float* __restrict__ bfp,
    _Float16* __restrict__ gru_pk, _Float16* __restrict__ wf_pk,
    float* __restrict__ Wf, float* __restrict__ bf)
{
  const int tid = threadIdx.x;
  __shared__ float Wf_s[128];

  // Wfp = proj_W * enc_W  (32x8), bfp = proj_W*enc_b + proj_b
  for (int e = tid; e < 256; e += 512) {
    int o = e >> 3, i = e & 7;
    float acc = 0.f;
    for (int k = 0; k < 32; ++k) acc += proj_W[o * 32 + k] * enc_W[k * 8 + i];
    Wfp[e] = acc;
  }
  if (tid < 32) {
    float acc = proj_b[tid];
    for (int k = 0; k < 32; ++k) acc += proj_W[tid * 32 + k] * enc_b[k];
    bfp[tid] = acc;
  }

  // GRU weights -> f16 B-fragments: [gate(3)][kc(2)][nt(2)][lane(32)][i(16)]
  for (int e = tid; e < 6144; e += 512) {
    int i = e & 15, l = (e >> 4) & 31, nt = (e >> 9) & 1, kc = (e >> 10) & 1, g = e >> 11;
    const float* W = (g == 0) ? Wz_W : (g == 1) ? Wr_W : Wh_W;
    int n = nt * 16 + (l & 15);
    int k = kc * 32 + ((i >= 8) ? 16 : 0) + (l >> 4) * 8 + (i & 7);
    gru_pk[e] = (_Float16)W[n * 64 + k];
  }

  // Wf = head_W * exp_W * comp_W  (4x32); bf = fused bias
  for (int e = tid; e < 128; e += 512) {
    int o = e >> 5, d = e & 31;
    float acc = 0.f;
    for (int j = 0; j < 32; ++j) {
      float hw = head_W[o * 32 + j];
      for (int k = 0; k < 16; ++k) acc += hw * exp_W[j * 16 + k] * comp_W[k * 32 + d];
    }
    Wf_s[e] = acc;
    Wf[e] = acc;
  }
  if (tid < 4) {
    float acc = head_b[tid];
    for (int j = 0; j < 32; ++j) {
      float t = exp_b[j];
      for (int k = 0; k < 16; ++k) t += exp_W[j * 16 + k] * comp_b[k];
      acc += head_W[tid * 32 + j] * t;
    }
    bf[tid] = acc;
  }
  __syncthreads();
  // Wf as f16 B-fragment, N padded 4 -> 16 with zeros
  for (int e = tid; e < 512; e += 512) {
    int i = e & 15, l = e >> 4, n = l & 15;
    wf_pk[e] = (n < 4) ? (_Float16)Wf_s[n * 32 + ((i >= 8) ? 16 : 0) + (l >> 4) * 8 + (i & 7)] : (_Float16)0.f;
  }
}

// ===========================================================================
// Encode: h = x @ Wfp^T + bfp, written directly in GRU A-fragment f16 order.
// A-frag (16-bit, 16x32): lane l, elem i -> M = l&15,
//   K = (i>=8?16:0) + (l>=16?8:0) + (i&7)
// ===========================================================================
__global__ __launch_bounds__(256) void encode_kernel(
    const float* __restrict__ x, const float* __restrict__ Wfp,
    const float* __restrict__ bfp, _Float16* __restrict__ h_pk)
{
  size_t gid = (size_t)blockIdx.x * 256 + threadIdx.x;  // 4*4096*32 lanes
  int l = gid & 31;
  size_t rest = gid >> 5;
  int t = (int)(rest & 4095);
  int bt = (int)(rest >> 12);
  int col = l & 15, hb = l >> 4;

  const float* xr = x + (((size_t)(bt * 16 + col) * TT) + t) * 8;
  float xv[8];
#pragma unroll
  for (int k = 0; k < 8; ++k) xv[k] = xr[k];

  v16h out;
#pragma unroll
  for (int i = 0; i < 16; ++i) {
    int K = ((i >= 8) ? 16 : 0) + hb * 8 + (i & 7);
    float acc = bfp[K];
#pragma unroll
    for (int k = 0; k < 8; ++k) acc += Wfp[K * 8 + k] * xv[k];
    out[i] = (_Float16)acc;
  }
  *(v16h*)(h_pk + (gid << 4)) = out;
}

// ===========================================================================
// GRU scan: 4 persistent single-wave workgroups, 16 batch rows each.
// Weights live in VGPRs as B-frags; 12 v_wmma per step; LDS transposes for
// the C-layout -> A-layout hidden-state feedback. Activations use
// v_exp/v_rcp short chains (critical path: 4096 dependent steps).
// ===========================================================================
__global__ __launch_bounds__(32) void gru_kernel(
    const _Float16* __restrict__ h_pk, const _Float16* __restrict__ gru_pk,
    const float* __restrict__ bz, const float* __restrict__ br,
    const float* __restrict__ bhv,
    float* __restrict__ states_pk, float* __restrict__ finals)
{
  const int bt = blockIdx.x;
  const int l = threadIdx.x;
  const int col = l & 15;
  const int hb = l >> 4;
  const int k0 = hb * 8;

  __shared__ float hc_lds[16][34];
  __shared__ float rh_lds[16][34];
#pragma unroll
  for (int r = 0; r < 16; ++r) hc_lds[r][l] = 0.f;

  v16h Wz[2][2], Wr[2][2], Wh[2][2];
#pragma unroll
  for (int kc = 0; kc < 2; ++kc)
#pragma unroll
    for (int nt = 0; nt < 2; ++nt) {
      Wz[kc][nt] = *(const v16h*)(gru_pk + (((size_t)((0 * 2 + kc) * 2 + nt) * 32 + l) << 4));
      Wr[kc][nt] = *(const v16h*)(gru_pk + (((size_t)((1 * 2 + kc) * 2 + nt) * 32 + l) << 4));
      Wh[kc][nt] = *(const v16h*)(gru_pk + (((size_t)((2 * 2 + kc) * 2 + nt) * 32 + l) << 4));
    }

  const float bz0 = bz[col], bz1 = bz[16 + col];
  const float br0 = br[col], br1 = br[16 + col];
  const float bh0 = bhv[col], bh1 = bhv[16 + col];

  v8f hc0, hc1;
#pragma unroll
  for (int r = 0; r < 8; ++r) { hc0[r] = 0.f; hc1[r] = 0.f; }
  __syncthreads();

  const size_t base = (size_t)bt * TT * 32;
  for (int t = 0; t < TT; ++t) {
    v16h ax = *(const v16h*)(h_pk + ((base + (size_t)t * 32 + l) << 4));
    if (t + 1 < TT)
      __builtin_prefetch(h_pk + ((base + (size_t)(t + 1) * 32 + l) << 4), 0, 3);

    v16h ahc;
#pragma unroll
    for (int i = 0; i < 8; ++i) {
      ahc[i]     = (_Float16)hc_lds[col][k0 + i];
      ahc[i + 8] = (_Float16)hc_lds[col][16 + k0 + i];
    }

    v8f z0, z1, r0, r1;
#pragma unroll
    for (int r = 0; r < 8; ++r) { z0[r] = bz0; z1[r] = bz1; r0[r] = br0; r1[r] = br1; }
    z0 = wmma_f16(ax, Wz[0][0], z0); z0 = wmma_f16(ahc, Wz[1][0], z0);
    z1 = wmma_f16(ax, Wz[0][1], z1); z1 = wmma_f16(ahc, Wz[1][1], z1);
    r0 = wmma_f16(ax, Wr[0][0], r0); r0 = wmma_f16(ahc, Wr[1][0], r0);
    r1 = wmma_f16(ax, Wr[0][1], r1); r1 = wmma_f16(ahc, Wr[1][1], r1);

#pragma unroll
    for (int r = 0; r < 8; ++r) {
      z0[r] = sigm(z0[r]);
      z1[r] = sigm(z1[r]);
      rh_lds[r + k0][col]      = sigm(r0[r]) * hc0[r];
      rh_lds[r + k0][16 + col] = sigm(r1[r]) * hc1[r];
    }
    __syncthreads();

    v16h arh;
#pragma unroll
    for (int i = 0; i < 8; ++i) {
      arh[i]     = (_Float16)rh_lds[col][k0 + i];
      arh[i + 8] = (_Float16)rh_lds[col][16 + k0 + i];
    }

    v8f c0, c1;
#pragma unroll
    for (int r = 0; r < 8; ++r) { c0[r] = bh0; c1[r] = bh1; }
    c0 = wmma_f16(ax, Wh[0][0], c0); c0 = wmma_f16(arh, Wh[1][0], c0);
    c1 = wmma_f16(ax, Wh[0][1], c1); c1 = wmma_f16(arh, Wh[1][1], c1);

    v8f hn0, hn1;
#pragma unroll
    for (int r = 0; r < 8; ++r) {
      hn0[r] = (1.f - z0[r]) * hc0[r] + z0[r] * fast_tanh(c0[r]);
      hn1[r] = (1.f - z1[r]) * hc1[r] + z1[r] * fast_tanh(c1[r]);
    }
    hc0 = hn0; hc1 = hn1;

    __syncthreads();
#pragma unroll
    for (int r = 0; r < 8; ++r) {
      hc_lds[r + k0][col]      = hn0[r];
      hc_lds[r + k0][16 + col] = hn1[r];
    }
    __syncthreads();

    float* sp = states_pk + ((base + (size_t)t * 32 + l) << 4);
    *(v8f*)sp = hn0;
    *(v8f*)(sp + 8) = hn1;

    if ((t & 3) == 3) {
      const int c = t >> 2;
#pragma unroll
      for (int r = 0; r < 8; ++r) {
        const size_t b = (size_t)(bt * 16 + r + k0);
        finals[(b * NC + c) * 32 + col]      = hn0[r];
        finals[(b * NC + c) * 32 + 16 + col] = hn1[r];
      }
    }
  }
}

// ===========================================================================
// Refine (2 fixed-point steps) + residual combine -> summ. Per-row VALU,
// uniform weight addresses -> scalar loads.
// ===========================================================================
__global__ __launch_bounds__(64) void refine_kernel(
    const float* __restrict__ finals,
    const float* __restrict__ SfS, const float* __restrict__ SfM,
    const float* __restrict__ SfD,
    const float* __restrict__ Mz_W, const float* __restrict__ Mz_b,
    const float* __restrict__ Mr_W, const float* __restrict__ Mr_b,
    const float* __restrict__ Mh_W, const float* __restrict__ Mh_b,
    const float* __restrict__ DfS, const float* __restrict__ DfM,
    const float* __restrict__ DfD,
    const float* __restrict__ phiS_W, const float* __restrict__ phiS_b,
    const float* __restrict__ phiD_W, const float* __restrict__ phiD_b,
    const float* __restrict__ res, float* __restrict__ summ)
{
  int idx = blockIdx.x * 64 + threadIdx.x;
  if (idx >= BB * (NC - 1)) return;
  int b = idx / (NC - 1), j = idx - b * (NC - 1);

  const float* hp = finals + ((size_t)b * NC + j) * 32;
  const float* hq = hp + 32;
  float S[32], M[32], D[32];
  for (int d = 0; d < 32; ++d) { S[d] = 0.5f * (hp[d] + hq[d]); M[d] = hp[d]; D[d] = hq[d]; }

  for (int step = 0; step < 2; ++step) {
    float Sn[32];
    for (int o = 0; o < 32; ++o) {
      float g = phiS_b[o], tv = 0.f;
      for (int k = 0; k < 32; ++k) {
        g  += phiS_W[o * 64 + k] * M[k] + phiS_W[o * 64 + 32 + k] * D[k];
        tv += SfS[o * 32 + k] * S[k] + SfM[o * 32 + k] * M[k] + SfD[o * 32 + k] * D[k];
      }
      Sn[o] = fast_tanh(tv) * sigm(g);
    }
    float z[32], r[32];
    for (int o = 0; o < 32; ++o) {
      float az = Mz_b[o], ar = Mr_b[o];
      for (int k = 0; k < 32; ++k) {
        az += Mz_W[o * 96 + k] * Sn[k] + Mz_W[o * 96 + 32 + k] * M[k] + Mz_W[o * 96 + 64 + k] * D[k];
        ar += Mr_W[o * 96 + k] * Sn[k] + Mr_W[o * 96 + 32 + k] * M[k] + Mr_W[o * 96 + 64 + k] * D[k];
      }
      z[o] = sigm(az); r[o] = sigm(ar);
    }
    float Mn[32];
    for (int o = 0; o < 32; ++o) {
      float ah = Mh_b[o];
      for (int k = 0; k < 32; ++k)
        ah += Mh_W[o * 96 + k] * Sn[k] + Mh_W[o * 96 + 32 + k] * (r[k] * M[k]) + Mh_W[o * 96 + 64 + k] * D[k];
      Mn[o] = (1.f - z[o]) * M[o] + z[o] * fast_tanh(ah);
    }
    float Dn[32];
    for (int o = 0; o < 32; ++o) {
      float g = phiD_b[o], tv = 0.f;
      for (int k = 0; k < 32; ++k) {
        g  += phiD_W[o * 64 + k] * Sn[k] + phiD_W[o * 64 + 32 + k] * Mn[k];
        tv += DfS[o * 32 + k] * Sn[k] + DfM[o * 32 + k] * Mn[k] + DfD[o * 32 + k] * D[k];
      }
      Dn[o] = fast_tanh(tv) * sigm(g);
    }
    for (int d = 0; d < 32; ++d) { S[d] = Sn[d]; M[d] = Mn[d]; D[d] = Dn[d]; }
  }

  const float rv = res[0];
  float* outp = summ + ((size_t)b * NC + j + 1) * 32;
  for (int d = 0; d < 32; ++d) outp[d] = M[d] + rv * (S[d] + D[d]);
  if (j == 0) {
    const float* f0 = finals + (size_t)b * NC * 32;
    float* s0 = summ + (size_t)b * NC * 32;
    for (int d = 0; d < 32; ++d) s0[d] = f0[d];
  }
}

// ===========================================================================
// QKV projection (per-row matvec, memory-trivial)
// ===========================================================================
__global__ __launch_bounds__(128) void qkv_kernel(
    const float* __restrict__ summ, const float* __restrict__ qkv_W,
    const float* __restrict__ qkv_b, float* __restrict__ qkvb)
{
  int row = blockIdx.x * 128 + threadIdx.x;
  if (row >= BB * NC) return;
  float in[32];
  const float* sr = summ + (size_t)row * 32;
  for (int k = 0; k < 32; ++k) in[k] = sr[k];
  float* qr = qkvb + (size_t)row * 96;
  for (int o = 0; o < 96; ++o) {
    float acc = qkv_b[o];
    for (int k = 0; k < 32; ++k) acc += qkv_W[o * 32 + k] * in[k];
    qr[o] = acc;
  }
}

// ===========================================================================
// Flash attention: 1 wave per (b, head, 16-query tile); per 16-key tile:
// one wmma for QK^T (K padded 16->32), online softmax via shfl_xor within
// 16-lane halves, one wmma for P@V. nc=1024 -> 64 key tiles per wave.
// ===========================================================================
__global__ __launch_bounds__(32) void attn_kernel(
    const float* __restrict__ qkvb, float* __restrict__ att)
{
  const int qt = blockIdx.x, h = blockIdx.y, b = blockIdx.z;
  const int l = threadIdx.x, col = l & 15, hb = l >> 4, k0 = hb * 8;
  __shared__ float p_lds[16][17];

  v16h aq;
#pragma unroll
  for (int i = 0; i < 8; ++i) {
    aq[i] = (_Float16)(0.25f *
        qkvb[((size_t)(b * NC + qt * 16 + col)) * 96 + h * 16 + k0 + i]);
    aq[i + 8] = (_Float16)0.f;
  }

  float m[8], se[8];
  v8f o;
#pragma unroll
  for (int r = 0; r < 8; ++r) { m[r] = -3.0e38f; se[r] = 0.f; o[r] = 0.f; }

  for (int kt = 0; kt < 64; ++kt) {
    v16h bk;
#pragma unroll
    for (int i = 0; i < 16; ++i)
      bk[i] = (i < 8)
          ? (_Float16)qkvb[((size_t)(b * NC + kt * 16 + col)) * 96 + 32 + h * 16 + k0 + i]
          : (_Float16)0.f;

    v8f s;
#pragma unroll
    for (int r = 0; r < 8; ++r) s[r] = 0.f;
    s = wmma_f16(aq, bk, s);

#pragma unroll
    for (int r = 0; r < 8; ++r) {
      float sv = s[r];
      float mx = sv;
#pragma unroll
      for (int msk = 8; msk >= 1; msk >>= 1) mx = fmaxf(mx, __shfl_xor(mx, msk, 32));
      float mnew = fmaxf(m[r], mx);
      float corr = __expf(m[r] - mnew);
      float p = __expf(sv - mnew);
      float ps = p;
#pragma unroll
      for (int msk = 8; msk >= 1; msk >>= 1) ps += __shfl_xor(ps, msk, 32);
      se[r] = se[r] * corr + ps;
      m[r] = mnew;
      o[r] *= corr;
      p_lds[r + k0][col] = p;   // P in C-layout -> LDS
    }
    __syncthreads();

    v16h ap;  // P as A-frag (K = key index, padded 16->32)
#pragma unroll
    for (int i = 0; i < 8; ++i) {
      ap[i] = (_Float16)p_lds[col][k0 + i];
      ap[i + 8] = (_Float16)0.f;
    }
    __syncthreads();

    v16h bv;
#pragma unroll
    for (int i = 0; i < 16; ++i)
      bv[i] = (i < 8)
          ? (_Float16)qkvb[((size_t)(b * NC + kt * 16 + k0 + i)) * 96 + 64 + h * 16 + col]
          : (_Float16)0.f;
    o = wmma_f16(ap, bv, o);
  }

#pragma unroll
  for (int r = 0; r < 8; ++r)
    att[((size_t)(b * NC + qt * 16 + r + k0)) * 32 + h * 16 + col] =
        o[r] * fast_rcp(se[r]);
}

// ===========================================================================
// Post: out-proj + LN + exact-GELU FFN, then fold xg through the fused
// output affine:  xgf[b][c][o] = bf[o] + sum_d xg[d] * Wf[o][d]
// ===========================================================================
__global__ __launch_bounds__(64) void post_kernel(
    const float* __restrict__ summ, const float* __restrict__ att,
    const float* __restrict__ out_W, const float* __restrict__ out_b,
    const float* __restrict__ ln_g, const float* __restrict__ ln_b,
    const float* __restrict__ f1_W, const float* __restrict__ f1_b,
    const float* __restrict__ f2_W, const float* __restrict__ f2_b,
    const float* __restrict__ Wf, const float* __restrict__ bf,
    float* __restrict__ xgf)
{
  int row = blockIdx.x * 64 + threadIdx.x;
  if (row >= BB * NC) return;
  const float* sr = summ + (size_t)row * 32;
  const float* ar = att + (size_t)row * 32;

  float xg[32];
  for (int o = 0; o < 32; ++o) {
    float acc = out_b[o];
    for (int k = 0; k < 32; ++k) acc += out_W[o * 32 + k] * ar[k];
    xg[o] = sr[o] + acc;
  }
  float mu = 0.f;
  for (int o = 0; o < 32; ++o) mu += xg[o];
  mu *= (1.f / 32.f);
  float var = 0.f;
  for (int o = 0; o < 32; ++o) { float d = xg[o] - mu; var += d * d; }
  var *= (1.f / 32.f);
  float inv = rsqrtf(var + 1e-5f);
  for (int o = 0; o < 32; ++o) xg[o] = (xg[o] - mu) * inv * ln_g[o] + ln_b[o];

  float f1o[64];
  for (int o = 0; o < 64; ++o) {
    float acc = f1_b[o];
    for (int k = 0; k < 32; ++k) acc += f1_W[o * 32 + k] * xg[k];
    f1o[o] = 0.5f * acc * (1.f + erff(acc * 0.70710678f));  // exact GELU
  }
  float xg2[32];
  for (int o = 0; o < 32; ++o) {
    float acc = f2_b[o];
    for (int k = 0; k < 64; ++k) acc += f2_W[o * 64 + k] * f1o[k];
    xg2[o] = xg[o] + acc;
  }
  for (int o = 0; o < 4; ++o) {
    float acc = bf[o];
    for (int d = 0; d < 32; ++d) acc += Wf[o * 32 + d] * xg2[d];
    xgf[(size_t)row * 4 + o] = acc;
  }
}

// ===========================================================================
// Final: y[b][t][o] = states[b][t][:] . Wf[o][:] + xgf[b][t/4][o]
// One wave per (bt, t): coalesced read of packed states, LDS detranspose,
// one wmma against Wf B-frag (N padded 4->16).
// ===========================================================================
__global__ __launch_bounds__(32) void final_kernel(
    const float* __restrict__ states_pk, const _Float16* __restrict__ wf_pk,
    const float* __restrict__ xgf, float* __restrict__ y)
{
  const int bid = blockIdx.x;              // bt*4096 + t
  const int bt = bid >> 12, t = bid & 4095;
  const int l = threadIdx.x, col = l & 15, hb = l >> 4, k0 = hb * 8;
  __shared__ float s_lds[16][34];

  const float* sp = states_pk + (((size_t)bid) * 32 + l) * 16;
#pragma unroll
  for (int i = 0; i < 16; ++i) {
    int tl = i >> 3, r = i & 7;
    s_lds[r + k0][tl * 16 + col] = sp[i];  // (batch_row, d)
  }
  __syncthreads();

  v16h a;
#pragma unroll
  for (int i = 0; i < 8; ++i) {
    a[i]     = (_Float16)s_lds[col][k0 + i];
    a[i + 8] = (_Float16)s_lds[col][16 + k0 + i];
  }
  v16h bfrag = *(const v16h*)(wf_pk + (size_t)l * 16);

  v8f acc;
#pragma unroll
  for (int r = 0; r < 8; ++r) acc[r] = 0.f;
  acc = wmma_f16(a, bfrag, acc);

  if (col < 4) {
    const int c = t >> 2;
#pragma unroll
    for (int r = 0; r < 8; ++r) {
      size_t b = (size_t)(bt * 16 + r + k0);
      y[(b * TT + t) * 4 + col] = acc[r] + xgf[(b * NC + c) * 4 + col];
    }
  }
}

// ===========================================================================
// Host orchestration (graph-capture safe: launches only)
// ===========================================================================
extern "C" void kernel_launch(void* const* d_in, const int* in_sizes, int n_in,
                              void* d_out, int out_size, void* d_ws, size_t ws_size,
                              hipStream_t stream) {
  (void)in_sizes; (void)n_in; (void)out_size; (void)ws_size;
  const float* P[44];
  for (int i = 0; i < 44; ++i) P[i] = (const float*)d_in[i];

  char* ws = (char*)d_ws;
  _Float16* h_pk    = (_Float16*)(ws + OFF_HPK);
  float* states_pk  = (float*)(ws + OFF_STATES);
  float* finals     = (float*)(ws + OFF_FINALS);
  float* summ       = (float*)(ws + OFF_SUMM);
  float* qkvb       = (float*)(ws + OFF_QKV);
  float* attb       = (float*)(ws + OFF_ATT);
  float* xgf        = (float*)(ws + OFF_XGF);
  float* Wfp        = (float*)(ws + OFF_WFP);
  float* bfp        = (float*)(ws + OFF_BFP);
  _Float16* gru_pk  = (_Float16*)(ws + OFF_GRUPK);
  _Float16* wf_pk   = (_Float16*)(ws + OFF_WFPK);
  float* Wf         = (float*)(ws + OFF_WF);
  float* bf         = (float*)(ws + OFF_BF);

  prep_kernel<<<1, 512, 0, stream>>>(
      P[IN_ENC_W], P[IN_ENC_B], P[IN_PROJ_W], P[IN_PROJ_B],
      P[IN_WZ_W], P[IN_WR_W], P[IN_WH_W],
      P[IN_COMP_W], P[IN_COMP_B], P[IN_EXP_W], P[IN_EXP_B],
      P[IN_HEAD_W], P[IN_HEAD_B],
      Wfp, bfp, gru_pk, wf_pk, Wf, bf);

  encode_kernel<<<2048, 256, 0, stream>>>(P[IN_X], Wfp, bfp, h_pk);

  gru_kernel<<<NBT, 32, 0, stream>>>(h_pk, gru_pk,
      P[IN_WZ_B], P[IN_WR_B], P[IN_WH_B], states_pk, finals);

  refine_kernel<<<(BB * (NC - 1)) / 64, 64, 0, stream>>>(
      finals, P[IN_SFS], P[IN_SFM], P[IN_SFD],
      P[IN_MZ_W], P[IN_MZ_B], P[IN_MR_W], P[IN_MR_B], P[IN_MH_W], P[IN_MH_B],
      P[IN_DFS], P[IN_DFM], P[IN_DFD],
      P[IN_PHIS_W], P[IN_PHIS_B], P[IN_PHID_W], P[IN_PHID_B],
      P[IN_RES], summ);

  qkv_kernel<<<(BB * NC) / 128, 128, 0, stream>>>(summ, P[IN_QKV_W], P[IN_QKV_B], qkvb);

  attn_kernel<<<dim3(NC / 16, 2, BB), 32, 0, stream>>>(qkvb, attb);

  post_kernel<<<(BB * NC) / 64, 64, 0, stream>>>(
      summ, attb, P[IN_OUT_W], P[IN_OUT_B], P[IN_LN_G], P[IN_LN_B],
      P[IN_F1_W], P[IN_F1_B], P[IN_F2_W], P[IN_F2_B], Wf, bf, xgf);

  final_kernel<<<NBT * TT, 32, 0, stream>>>(states_pk, wf_pk, xgf, (float*)d_out);
}
